// MultiHeadAttention_9156870275103
// MI455X (gfx1250) — compile-verified, matrix operand order checked by reading, not run on hardware
//
#include <hip/hip_runtime.h>
#include <hip/hip_bf16.h>

// MHA forward for B=2, S=2048, D=1024, H=16, Dh=64 on gfx1250 (wave32, WMMA).
#define SB 2
#define SS 2048
#define SD 1024
#define SH 16
#define SHD 64

typedef __attribute__((ext_vector_type(16))) __bf16 v16bf;
typedef __attribute__((ext_vector_type(8)))  float  v8f;

static __device__ __forceinline__ v8f wmma_bf16(v16bf a, v16bf b, v8f c) {
  // D = A(16x32) x B(32x16) + C, fp32 accumulate
  return __builtin_amdgcn_wmma_f32_16x16x32_bf16(false, a, false, b, (short)0, c, false, false);
}

// Load a 16x32 bf16 A/B fragment from a row-major tile (row stride in elems).
// ISA 7.12.2: lane<16 -> row=lane, K={0..7,16..23}; lane>=16 -> row=lane-16, K={8..15,24..31}.
static __device__ __forceinline__ v16bf load_frag(const __bf16* base, int stride, int lane) {
  const int m  = lane & 15;
  const int kb = (lane >> 4) << 3;   // 0 or 8
  const __bf16* p = base + (size_t)m * stride + kb;
  v16bf f;
#pragma unroll
  for (int j = 0; j < 8; ++j) { f[j] = p[j]; f[8 + j] = p[16 + j]; }
  return f;
}

// CDNA5 async global->LDS copy (16B per lane) + wait, via inline asm
// (portable across ROCm7.2 / amdgpu-toolchain builtin arity differences).
static __device__ __forceinline__ void async_copy_b128(unsigned int lds_off, const void* gptr) {
  asm volatile("global_load_async_to_lds_b128 %0, %1, off"
               :: "v"(lds_off), "v"((unsigned long long)(uintptr_t)gptr) : "memory");
}
static __device__ __forceinline__ void wait_async0() {
  asm volatile("s_wait_asynccnt 0x0" ::: "memory");
}

// ---------------------------------------------------------------------------
// Kernel 1: Y = X @ W^T + b  (X fp32 [4096,1024], W fp32 [1024,1024])
// Output bf16, head-major [B,H,S,64]. Block tile 128x128; wave tile 32x64.
// ---------------------------------------------------------------------------
__global__ void qkv_proj_kernel(const float* __restrict__ X, const float* __restrict__ W,
                                const float* __restrict__ bias, __bf16* __restrict__ Yh) {
  __shared__ __bf16 As[128][32];
  __shared__ __bf16 Ws[128][32];
  const int tid  = threadIdx.x;
  const int w    = tid >> 5;
  const int lane = tid & 31;
  const int wy   = w & 3;           // 4 row-groups of 32
  const int wx   = w >> 2;          // 2 col-groups of 64
  const int row0 = blockIdx.x * 128;
  const int col0 = blockIdx.y * 128;
  v8f acc[2][4] = {};

  for (int kc = 0; kc < SD; kc += 32) {
    __syncthreads();
#pragma unroll
    for (int e = 0; e < 16; ++e) {           // 128x32 slabs, 16 elems/thread each
      int idx = tid * 16 + e;
      int r = idx >> 5, c = idx & 31;
      As[r][c] = (__bf16)X[(size_t)(row0 + r) * SD + kc + c];
      Ws[r][c] = (__bf16)W[(size_t)(col0 + r) * SD + kc + c];
    }
    __syncthreads();

    v16bf a0 = load_frag(&As[wy * 32][0], 32, lane);
    v16bf a1 = load_frag(&As[wy * 32 + 16][0], 32, lane);
#pragma unroll
    for (int nt = 0; nt < 4; ++nt) {
      v16bf b = load_frag(&Ws[wx * 64 + nt * 16][0], 32, lane);
      acc[0][nt] = wmma_bf16(a0, b, acc[0][nt]);
      acc[1][nt] = wmma_bf16(a1, b, acc[1][nt]);
    }
  }

  const int n = lane & 15, mh = lane >> 4;
#pragma unroll
  for (int mi = 0; mi < 2; ++mi) {
#pragma unroll
    for (int r = 0; r < 8; ++r) {
      const int m    = mi * 16 + r + (mh << 3);
      const int rowg = row0 + wy * 32 + m;       // global row in [0, B*S)
      const int bb   = rowg >> 11;               // / 2048
      const int s    = rowg & (SS - 1);
#pragma unroll
      for (int nt = 0; nt < 4; ++nt) {
        const int ng = col0 + wx * 64 + nt * 16 + n;   // global col in [0,1024)
        const int h  = ng >> 6, d = ng & 63;
        float v = acc[mi][nt][r] + bias[ng];
        Yh[(((size_t)bb * SH + h) * SS + s) * SHD + d] = (__bf16)v;
      }
    }
  }
}

// ---------------------------------------------------------------------------
// Kernel 2: scores = (q @ k^T)/8, causal mask, softmax -> prob (fp32, d_out)
// One block per (b, h, 32-row q band). Scores staged bf16 in 128KB LDS
// (within the 320KB WGP budget). 4 WMMAs per 2 k-fragment loads.
// ---------------------------------------------------------------------------
__global__ void attn_kernel(const __bf16* __restrict__ qbf, const __bf16* __restrict__ kbf,
                            float* __restrict__ prob) {
  __shared__ __bf16 sc[32][SS];                // 128 KB
  const int tid  = threadIdx.x;
  const int w    = tid >> 5;
  const int lane = tid & 31;
  const int b = blockIdx.z, h = blockIdx.y, qb = blockIdx.x;
  const size_t hoff = ((size_t)b * SH + h) * SS * SHD;
  const __bf16* qh = qbf + hoff;
  const __bf16* kh = kbf + hoff;
  const int q0 = qb * 32;

  const v16bf aq00 = load_frag(qh + (size_t)q0 * SHD, SHD, lane);
  const v16bf aq01 = load_frag(qh + (size_t)q0 * SHD + 32, SHD, lane);
  const v16bf aq10 = load_frag(qh + (size_t)(q0 + 16) * SHD, SHD, lane);
  const v16bf aq11 = load_frag(qh + (size_t)(q0 + 16) * SHD + 32, SHD, lane);
  const int n = lane & 15, mh = lane >> 4;

  for (int t = 0; t < 16; ++t) {               // wave w covers col tiles [16w,16w+16)
    const int c0 = (w * 16 + t) * 16;
    v16bf bk0 = load_frag(kh + (size_t)c0 * SHD, SHD, lane);
    v16bf bk1 = load_frag(kh + (size_t)c0 * SHD + 32, SHD, lane);
    v8f acc0 = {}, acc1 = {};
    acc0 = wmma_bf16(aq00, bk0, acc0);
    acc0 = wmma_bf16(aq01, bk1, acc0);
    acc1 = wmma_bf16(aq10, bk0, acc1);
    acc1 = wmma_bf16(aq11, bk1, acc1);
#pragma unroll
    for (int r = 0; r < 8; ++r) {
      const int m  = r + (mh << 3);
      const int cg = c0 + n;
      float v0 = acc0[r] * 0.125f;             // 1/sqrt(64)
      float v1 = acc1[r] * 0.125f;
      if (cg > q0 + m)      v0 = -1e9f;        // causal mask
      if (cg > q0 + 16 + m) v1 = -1e9f;
      sc[m][cg]      = (__bf16)v0;
      sc[16 + m][cg] = (__bf16)v1;
    }
  }
  __syncthreads();

  float* ph = prob + ((size_t)b * SH + h) * SS * SS;
#pragma unroll
  for (int rr = 0; rr < 4; ++rr) {             // 4 rows per wave
    const int r  = w * 4 + rr;
    const int qg = q0 + r;
    float mx = -3.0e38f;
    for (int c = lane; c < SS; c += 32) mx = fmaxf(mx, (float)sc[r][c]);
#pragma unroll
    for (int o = 16; o; o >>= 1) mx = fmaxf(mx, __shfl_xor(mx, o, 32));
    float sum = 0.f;
    for (int c = lane; c < SS; c += 32) sum += __expf((float)sc[r][c] - mx);
#pragma unroll
    for (int o = 16; o; o >>= 1) sum += __shfl_xor(sum, o, 32);
    const float inv = 1.0f / sum;
    float* pr = ph + (size_t)qg * SS;
    for (int c = lane; c < SS; c += 32) pr[c] = __expf((float)sc[r][c] - mx) * inv;
  }
}

// ---------------------------------------------------------------------------
// Kernel 3: context = prob @ v  ([2048x2048] fp32 x [2048x64] bf16) per head.
// Block = 256 prob rows of one head (32 rows/wave). V k-slab staged TRANSPOSED
// in LDS so B fragments use the contiguous ds_load_b128 path.
// Output bf16 [B,S,H*64].
// ---------------------------------------------------------------------------
__global__ void context_kernel(const float* __restrict__ prob, const __bf16* __restrict__ vbf,
                               __bf16* __restrict__ ctx) {
  __shared__ __bf16 vtT[SHD][32];              // [d][k] 4 KB
  const int tid  = threadIdx.x;
  const int w    = tid >> 5;
  const int lane = tid & 31;
  const int b = blockIdx.z, h = blockIdx.y;
  const int row0 = blockIdx.x * 256;
  const float*  ph = prob + ((size_t)b * SH + h) * SS * SS;
  const __bf16* vh = vbf  + ((size_t)b * SH + h) * SS * SHD;
  const int n = lane & 15, mh = lane >> 4, kb = mh << 3;
  v8f acc[2][4] = {};

  for (int kc = 0; kc < SS; kc += 32) {
    __syncthreads();
#pragma unroll
    for (int e = 0; e < 8; ++e) {              // 32x64 V slab -> transposed LDS
      int idx = tid * 8 + e;
      vtT[idx & 63][idx >> 6] = vh[(size_t)kc * SHD + idx];
    }
    __syncthreads();

    v16bf a[2];
#pragma unroll
    for (int mi = 0; mi < 2; ++mi) {           // prob rows -> bf16 A fragments
      const int arow = row0 + w * 32 + mi * 16 + n;
      const float* ap = ph + (size_t)arow * SS + kc + kb;
#pragma unroll
      for (int j = 0; j < 8; ++j) { a[mi][j] = (__bf16)ap[j]; a[mi][8 + j] = (__bf16)ap[16 + j]; }
    }
#pragma unroll
    for (int nt = 0; nt < 4; ++nt) {
      v16bf bv = load_frag(&vtT[nt * 16][0], 32, lane);
      acc[0][nt] = wmma_bf16(a[0], bv, acc[0][nt]);
      acc[1][nt] = wmma_bf16(a[1], bv, acc[1][nt]);
    }
  }

#pragma unroll
  for (int mi = 0; mi < 2; ++mi) {
#pragma unroll
    for (int r = 0; r < 8; ++r) {
      const int m = mi * 16 + r + (mh << 3);
      const int s = row0 + w * 32 + m;
#pragma unroll
      for (int nt = 0; nt < 4; ++nt) {
        const int d = nt * 16 + n;
        ctx[((size_t)b * SS + s) * SD + h * SHD + d] = (__bf16)acc[mi][nt][r];
      }
    }
  }
}

// ---------------------------------------------------------------------------
// Kernel 4: out = ctx @ Wo^T + bo  (ctx bf16 [4096,1024], out fp32 to d_out)
// Block tile 128x128, wave tile 32x64. The bf16 A slab is a pure copy, staged
// with CDNA5 async global->LDS (ASYNCcnt) instead of load+store.
// ---------------------------------------------------------------------------
__global__ void out_proj_kernel(const __bf16* __restrict__ ctx, const float* __restrict__ W,
                                const float* __restrict__ bias, float* __restrict__ out) {
  __shared__ __bf16 As[128][32];
  __shared__ __bf16 Ws[128][32];
  const int tid  = threadIdx.x;
  const int w    = tid >> 5;
  const int lane = tid & 31;
  const int wy   = w & 3;
  const int wx   = w >> 2;
  const int row0 = blockIdx.x * 128;
  const int col0 = blockIdx.y * 128;
  const unsigned int asBase = (unsigned int)(uintptr_t)&As[0][0];
  v8f acc[2][4] = {};

  for (int kc = 0; kc < SD; kc += 32) {
    __syncthreads();
    {                                          // async-copy 128x32 bf16 A slab: 1 x B128/thread
      const int r = tid >> 1, c = (tid & 1) * 16;
      async_copy_b128(asBase + (unsigned int)(r * 32 + c) * 2,
                      ctx + (size_t)(row0 + r) * SD + kc + c);
    }
#pragma unroll
    for (int e = 0; e < 16; ++e) {             // 128x32 W slab with fp32->bf16 cvt
      int idx = tid * 16 + e;
      int r = idx >> 5, c = idx & 31;
      Ws[r][c] = (__bf16)W[(size_t)(col0 + r) * SD + kc + c];
    }
    wait_async0();
    __syncthreads();

    v16bf a0 = load_frag(&As[wy * 32][0], 32, lane);
    v16bf a1 = load_frag(&As[wy * 32 + 16][0], 32, lane);
#pragma unroll
    for (int nt = 0; nt < 4; ++nt) {
      v16bf b = load_frag(&Ws[wx * 64 + nt * 16][0], 32, lane);
      acc[0][nt] = wmma_bf16(a0, b, acc[0][nt]);
      acc[1][nt] = wmma_bf16(a1, b, acc[1][nt]);
    }
  }

  const int n = lane & 15, mh = lane >> 4;
#pragma unroll
  for (int mi = 0; mi < 2; ++mi) {
#pragma unroll
    for (int r = 0; r < 8; ++r) {
      const int m    = mi * 16 + r + (mh << 3);
      const int rowg = row0 + wy * 32 + m;
#pragma unroll
      for (int nt = 0; nt < 4; ++nt) {
        const int ng = col0 + wx * 64 + nt * 16 + n;
        out[(size_t)rowg * SD + ng] = acc[mi][nt][r] + bias[ng];
      }
    }
  }
}

// ---------------------------------------------------------------------------
extern "C" void kernel_launch(void* const* d_in, const int* in_sizes, int n_in,
                              void* d_out, int out_size, void* d_ws, size_t ws_size,
                              hipStream_t stream) {
  const float* Q  = (const float*)d_in[0];
  const float* K  = (const float*)d_in[1];
  const float* V  = (const float*)d_in[2];
  // d_in[3] = attn_mask (causal; applied analytically via indices)
  const float* Wq = (const float*)d_in[4];
  const float* bq = (const float*)d_in[5];
  const float* Wk = (const float*)d_in[6];
  const float* bk = (const float*)d_in[7];
  const float* Wv = (const float*)d_in[8];
  const float* bv = (const float*)d_in[9];
  const float* Wo = (const float*)d_in[10];
  const float* bo = (const float*)d_in[11];

  float* out  = (float*)d_out;                       // [B,S,1024]  = 4,194,304 floats
  float* prob = out + (size_t)SB * SS * SD;          // [B,H,S,S]   = 134,217,728 floats

  const size_t PROJ = (size_t)SB * SS * SD;          // 4,194,304 elems
  __bf16* qbf = (__bf16*)d_ws;                       // bf16 head-major projections
  __bf16* kbf = qbf + PROJ;
  __bf16* vbf = kbf + PROJ;
  __bf16* ctx = vbf + PROJ;                          // bf16 merged-head context
  (void)in_sizes; (void)n_in; (void)out_size; (void)ws_size;  // 32 MB ws used

  dim3 blk(256);
  dim3 gproj(SB * SS / 128, SD / 128);               // 32 x 8
  qkv_proj_kernel<<<gproj, blk, 0, stream>>>(Q, Wq, bq, qbf);
  qkv_proj_kernel<<<gproj, blk, 0, stream>>>(K, Wk, bk, kbf);
  qkv_proj_kernel<<<gproj, blk, 0, stream>>>(V, Wv, bv, vbf);

  dim3 gattn(SS / 32, SH, SB);                       // 64 x 16 x 2
  attn_kernel<<<gattn, blk, 0, stream>>>(qbf, kbf, prob);

  dim3 gctx(SS / 256, SH, SB);                       // 8 x 16 x 2
  context_kernel<<<gctx, blk, 0, stream>>>(prob, vbf, ctx);

  out_proj_kernel<<<gproj, blk, 0, stream>>>(ctx, Wo, bo, out);
}
